// T2T_module_49976239456367
// MI455X (gfx1250) — matrix operand order repaired and checked by reading, not compiled, not run on hardware
//
#include <hip/hip_runtime.h>
#include <hip/hip_bf16.h>
#include <math.h>

// ---------------------------------------------------------------------------
// T2T-ViT stem (unfold + Performer x2 + final projection) for gfx1250.
// All dense GEMMs use v_wmma_f32_16x16x32_f16 (f16 A/B, f32 accum).
// wave32; one wave computes a 16x(16*NT) output strip, A fragment reused.
// Weights / performer operands are pre-packed to padded fp16 so all WMMA
// fragment loads are unguarded, aligned b128 vector loads.
// ---------------------------------------------------------------------------

typedef __attribute__((ext_vector_type(16))) _Float16 v16h;
typedef __attribute__((ext_vector_type(8)))  _Float16 v8h;
typedef __attribute__((ext_vector_type(8)))  float    v8f;

#define FULL32 32

// A fragment (16-bit A 16x32, ISA 7.12.2): lane<16 -> M=lane, K {kb..kb+7} u
// {kb+16..kb+23}, kb=0; lane>=16 -> M=lane-16, kb=8.  p points at (row, k0+kb).
__device__ __forceinline__ v16h ld_frag_a(const _Float16* p) {
    const v8h* q = (const v8h*)p;
    v8h lo = q[0];   // K offsets 0..7
    v8h hi = q[2];   // K offsets 16..23
    return __builtin_shufflevector(lo, hi, 0,1,2,3,4,5,6,7,8,9,10,11,12,13,14,15);
}

// B fragment: same K layout as A: K {kb..kb+7} u {kb+16..kb+23}, kb=8 for lane>=16.
// p points at (col, k0+kb), 16B aligned.
__device__ __forceinline__ v16h ld_frag_b(const _Float16* p) {
    const v8h* q = (const v8h*)p;
    v8h lo = q[0];
    v8h hi = q[2];
    return __builtin_shufflevector(lo, hi, 0,1,2,3,4,5,6,7,8,9,10,11,12,13,14,15);
}

// ---------------------------------------------------------------------------
// Weight convert: fp32 [N, Kd] -> fp16 [N, KdPad], zero padded.
// ---------------------------------------------------------------------------
__global__ void convert_w_kernel(const float* __restrict__ W, int N, int Kd, int KdPad,
                                 _Float16* __restrict__ W16)
{
    int i = blockIdx.x * 256 + threadIdx.x;
    if (i >= N * KdPad) return;
    int n = i / KdPad, k = i % KdPad;
    W16[i] = (k < Kd) ? (_Float16)W[(size_t)n * Kd + k] : (_Float16)0.f;
}

// ---------------------------------------------------------------------------
// Unfold (im2col, channel-major: d = c*k*k + ky*k + kx) + optional LayerNorm,
// writing zero-padded fp16 rows [tok, dPad]. One wave per token; patch cached
// in LDS so the gather happens once.
// srcTokenMajor=0: img is (B,C,H,W).  =1: img is [b*T + h*W+w, C].
// ---------------------------------------------------------------------------
__global__ void unfold_ln_kernel(const float* __restrict__ img, int srcTokenMajor,
                                 int C, int H, int W, int ksz, int stride, int pad,
                                 int Ht, int Wt, int d, int dPad,
                                 const float* __restrict__ g, const float* __restrict__ bvec,
                                 int doLN, _Float16* __restrict__ A)
{
    __shared__ float patch[576];     // max d
    const int tok  = blockIdx.x;
    const int T    = Ht * Wt;
    const int b    = tok / T;
    const int t    = tok % T;
    const int ph   = t / Wt, pw = t % Wt;
    const int lane = threadIdx.x;
    const int kk   = ksz * ksz;

    float s = 0.f, s2 = 0.f;
    for (int idx = lane; idx < d; idx += FULL32) {
        int c  = idx / kk;
        int r  = idx % kk;
        int ky = r / ksz, kx = r % ksz;
        int h  = ph * stride - pad + ky;
        int w  = pw * stride - pad + kx;
        float v = 0.f;
        if (h >= 0 && h < H && w >= 0 && w < W) {
            v = srcTokenMajor
                ? img[((size_t)b * H * W + (size_t)h * W + w) * C + c]
                : img[(((size_t)b * C + c) * H + h) * W + w];
        }
        patch[idx] = v;
        s += v; s2 += v * v;
    }
    for (int o = 16; o >= 1; o >>= 1) { s += __shfl_xor(s, o, FULL32); s2 += __shfl_xor(s2, o, FULL32); }
    const float invd = 1.f / (float)d;
    const float mu   = s * invd;
    const float var  = s2 * invd - mu * mu;
    const float rs   = rsqrtf(var + 1e-5f);

    _Float16* row = A + (size_t)tok * dPad;
    for (int idx = lane; idx < dPad; idx += FULL32) {
        float v = 0.f;
        if (idx < d) {
            v = patch[idx];
            if (doLN) v = (v - mu) * rs * g[idx] + bvec[idx];
        }
        row[idx] = (_Float16)v;
    }
}

// ---------------------------------------------------------------------------
// Row-GEMM: C[row,n] = sum_k A[row,k]*W16[n,k] + bias[n]
//   A   : fp16 [R, lda], zero-padded, lda mult of 32
//   W16 : fp16 [N, lda], zero-padded (preconverted weights)
// One wave computes 16 x (16*NT); A fragment reused across NT wmmas.
// grid = (R/16, N/(16*NT)).
// Fused: bias, optional exact GELU, optional residual, fp32/fp16 stores.
// ---------------------------------------------------------------------------
template <int NT>
__global__ void wmma_gemm_kernel(const _Float16* __restrict__ A, int lda,
                                 const _Float16* __restrict__ W16,
                                 const float* __restrict__ bias,
                                 const float* __restrict__ res, int resStride, int resOff,
                                 float* __restrict__ Cout, _Float16* __restrict__ C16,
                                 int N, int act)
{
    const int lane  = threadIdx.x;
    const int tileM = blockIdx.x;
    const int nBase = blockIdx.y * (16 * NT);

    const int rowA = tileM * 16 + (lane & 15);
    const int kbA  = (lane & 16) ? 8  : 0;
    const int kbB  = (lane & 16) ? 8 : 0;

    const _Float16* arow = A + (size_t)rowA * lda + kbA;
    const _Float16* wbase = W16 + (size_t)(nBase + (lane & 15)) * lda + kbB;

    v8f c[NT] = {};
    for (int k0 = 0; k0 < lda; k0 += 32) {
        v16h af = ld_frag_a(arow + k0);
        #pragma unroll
        for (int t = 0; t < NT; ++t) {
            v16h bf = ld_frag_b(wbase + (size_t)t * 16 * lda + k0);
            c[t] = __builtin_amdgcn_wmma_f32_16x16x32_f16(false, af, false, bf,
                                                          (short)0, c[t], false, false);
        }
    }

    #pragma unroll
    for (int t = 0; t < NT; ++t) {
        const int colOut = nBase + t * 16 + (lane & 15);
        const float bv = bias[colOut];
        #pragma unroll
        for (int j = 0; j < 8; ++j) {
            int rowOut = tileM * 16 + j + ((lane & 16) ? 8 : 0);
            float val = c[t][j] + bv;
            if (act == 1)  // exact GELU
                val = 0.5f * val * (1.f + erff(val * 0.70710678118654752f));
            if (res)
                val += res[(size_t)rowOut * resStride + resOff + colOut];
            if (Cout) Cout[(size_t)rowOut * N + colOut] = val;
            if (C16)  C16 [(size_t)rowOut * N + colOut] = (_Float16)val;
        }
    }
}

// ---------------------------------------------------------------------------
// prm_exp: kp[t,m] = exp(w[m,:]·x - 0.5*|x|^2)/sqrt(32); x = k or q slice.
// One wave per token; lane = m.
// ---------------------------------------------------------------------------
__global__ void prm_exp_kernel(const float* __restrict__ kqv,
                               const float* __restrict__ wmat,  // [32,64]
                               float* __restrict__ KP, float* __restrict__ QP)
{
    const int tok  = blockIdx.x;
    const int lane = threadIdx.x;
    const float* base = kqv + (size_t)tok * 192;
    const float* wr   = wmat + (size_t)lane * 64;

    for (int pass = 0; pass < 2; ++pass) {
        const float* xv = base + pass * 64;
        float x0 = xv[lane], x1 = xv[lane + 32];
        float xd = x0 * x0 + x1 * x1;
        for (int o = 16; o >= 1; o >>= 1) xd += __shfl_xor(xd, o, FULL32);
        xd *= 0.5f;
        float acc = 0.f;
        #pragma unroll 8
        for (int dd = 0; dd < 64; ++dd) acc += xv[dd] * wr[dd];
        float outv = expf(acc - xd) * 0.17677669529663689f;  // 1/sqrt(32)
        (pass ? QP : KP)[(size_t)tok * 32 + lane] = outv;
    }
}

// ---------------------------------------------------------------------------
// kps[b,m] = sum_t KP[b,t,m].  One block (8 groups of 32) per batch.
// ---------------------------------------------------------------------------
__global__ void kps_kernel(const float* __restrict__ KP, int T, float* __restrict__ KPS)
{
    __shared__ float red[256];
    const int b   = blockIdx.x;
    const int tid = threadIdx.x;
    const int m   = tid & 31;
    const int grp = tid >> 5;
    float acc = 0.f;
    for (int t = grp; t < T; t += 8)
        acc += KP[((size_t)b * T + t) * 32 + m];
    red[tid] = acc;
    __syncthreads();
    if (tid < 32) {
        float s = 0.f;
        #pragma unroll
        for (int g2 = 0; g2 < 8; ++g2) s += red[g2 * 32 + m];
        KPS[b * 32 + m] = s;
    }
}

// ---------------------------------------------------------------------------
// Pack v^T (fp16 [b, n(64), Tpad]) and kp^T (fp16 [b, m(32), Tpad]) with
// zero K-tail so kptv WMMA loads are unguarded b128s.
// ---------------------------------------------------------------------------
__global__ void pack_vt_kernel(const float* __restrict__ kqv, int T, int Tpad,
                               _Float16* __restrict__ vt)
{
    int i = blockIdx.x * 256 + threadIdx.x;
    if (i >= 64 * 64 * Tpad) return;                 // B * 64 * Tpad
    int t = i % Tpad, r = i / Tpad;
    int n = r & 63, b = r >> 6;
    float v = (t < T) ? kqv[((size_t)b * T + t) * 192 + 128 + n] : 0.f;
    vt[i] = (_Float16)v;
}

__global__ void pack_kpt_kernel(const float* __restrict__ KP, int T, int Tpad,
                                _Float16* __restrict__ kpt)
{
    int i = blockIdx.x * 256 + threadIdx.x;
    if (i >= 64 * 32 * Tpad) return;                 // B * 32 * Tpad
    int t = i % Tpad, r = i / Tpad;
    int m = r & 31, b = r >> 5;
    float v = (t < T) ? KP[((size_t)b * T + t) * 32 + m] : 0.f;
    kpt[i] = (_Float16)v;
}

// ---------------------------------------------------------------------------
// kptv[b,n,m] = sum_t v[b,t,n]*kp[b,t,m]  (64x32, K=Tpad) via WMMA.
// One block (8 waves) per batch: 4 n-tiles x 2 m-tiles.
// ---------------------------------------------------------------------------
__global__ void kptv_kernel(const _Float16* __restrict__ vt, const _Float16* __restrict__ kpt,
                            int Tpad, float* __restrict__ KPTV)
{
    const int b     = blockIdx.x;
    const int wave  = threadIdx.x >> 5;
    const int lane  = threadIdx.x & 31;
    const int mTile = wave & 3;   // n tiles (rows 0..63)
    const int nTile = wave >> 2;  // m tiles (cols 0..31)

    const int rowA = mTile * 16 + (lane & 15);
    const int kbA  = (lane & 16) ? 8  : 0;
    const int colB = nTile * 16 + (lane & 15);
    const int kbB  = (lane & 16) ? 8 : 0;

    const _Float16* arow = vt  + ((size_t)b * 64 + rowA) * Tpad + kbA;
    const _Float16* brow = kpt + ((size_t)b * 32 + colB) * Tpad + kbB;

    v8f c = {};
    for (int k0 = 0; k0 < Tpad; k0 += 32) {
        v16h af = ld_frag_a(arow + k0);
        v16h bf = ld_frag_b(brow + k0);
        c = __builtin_amdgcn_wmma_f32_16x16x32_f16(false, af, false, bf,
                                                   (short)0, c, false, false);
    }
    #pragma unroll
    for (int j = 0; j < 8; ++j) {
        int n = mTile * 16 + j + ((lane & 16) ? 8 : 0);
        int m = nTile * 16 + (lane & 15);
        KPTV[((size_t)b * 64 + n) * 32 + m] = c[j];
    }
}

// ---------------------------------------------------------------------------
// y_attn[t,n] = (sum_m qp[t,m]*kptv[b,n,m]) / (D[t]+eps), D = qp·kps.
// One wave per token. Output fp16 rows.
// ---------------------------------------------------------------------------
__global__ void yattn_kernel(const float* __restrict__ QP, const float* __restrict__ KPS,
                             const float* __restrict__ KPTV, int T,
                             _Float16* __restrict__ A)
{
    const int tok  = blockIdx.x;
    const int lane = threadIdx.x;
    const int b    = tok / T;
    const float qpv = QP[(size_t)tok * 32 + lane];
    float D = qpv * KPS[b * 32 + lane];
    for (int o = 16; o >= 1; o >>= 1) D += __shfl_xor(D, o, FULL32);
    const float inv = 1.f / (D + 1e-8f);

    #pragma unroll
    for (int half = 0; half < 2; ++half) {
        const int n = lane + half * 32;
        const float* kv = KPTV + ((size_t)b * 64 + n) * 32;
        float acc = 0.f;
        #pragma unroll
        for (int m = 0; m < 32; ++m) {
            float qm = __shfl(qpv, m, FULL32);
            acc += qm * kv[m];
        }
        A[(size_t)tok * 64 + n] = (_Float16)(acc * inv);
    }
}

// ---------------------------------------------------------------------------
// LayerNorm over 64 columns fp32 -> fp16 rows. One wave per token.
// ---------------------------------------------------------------------------
__global__ void ln64_kernel(const float* __restrict__ Y, const float* __restrict__ g,
                            const float* __restrict__ bvec, _Float16* __restrict__ A)
{
    const int tok  = blockIdx.x;
    const int lane = threadIdx.x;
    float x0 = Y[(size_t)tok * 64 + lane];
    float x1 = Y[(size_t)tok * 64 + lane + 32];
    float s = x0 + x1, s2 = x0 * x0 + x1 * x1;
    for (int o = 16; o >= 1; o >>= 1) { s += __shfl_xor(s, o, FULL32); s2 += __shfl_xor(s2, o, FULL32); }
    const float mu  = s * (1.f / 64.f);
    const float var = s2 * (1.f / 64.f) - mu * mu;
    const float rs  = rsqrtf(var + 1e-5f);
    A[(size_t)tok * 64 + lane]      = (_Float16)((x0 - mu) * rs * g[lane]      + bvec[lane]);
    A[(size_t)tok * 64 + lane + 32] = (_Float16)((x1 - mu) * rs * g[lane + 32] + bvec[lane + 32]);
}

// ---------------------------------------------------------------------------
// Host orchestration
// ---------------------------------------------------------------------------
extern "C" void kernel_launch(void* const* d_in, const int* in_sizes, int n_in,
                              void* d_out, int out_size, void* d_ws, size_t ws_size,
                              hipStream_t stream)
{
    (void)in_sizes; (void)n_in; (void)out_size; (void)ws_size;

    const float* x = (const float*)d_in[0];
    const float* p1_n1g = (const float*)d_in[1];
    const float* p1_n1b = (const float*)d_in[2];
    const float* p1_kqvw= (const float*)d_in[3];
    const float* p1_kqvb= (const float*)d_in[4];
    const float* p1_w   = (const float*)d_in[5];
    const float* p1_prw = (const float*)d_in[6];
    const float* p1_prb = (const float*)d_in[7];
    const float* p1_n2g = (const float*)d_in[8];
    const float* p1_n2b = (const float*)d_in[9];
    const float* p1_m1w = (const float*)d_in[10];
    const float* p1_m1b = (const float*)d_in[11];
    const float* p1_m2w = (const float*)d_in[12];
    const float* p1_m2b = (const float*)d_in[13];
    const float* p2_n1g = (const float*)d_in[14];
    const float* p2_n1b = (const float*)d_in[15];
    const float* p2_kqvw= (const float*)d_in[16];
    const float* p2_kqvb= (const float*)d_in[17];
    const float* p2_w   = (const float*)d_in[18];
    const float* p2_prw = (const float*)d_in[19];
    const float* p2_prb = (const float*)d_in[20];
    const float* p2_n2g = (const float*)d_in[21];
    const float* p2_n2b = (const float*)d_in[22];
    const float* p2_m1w = (const float*)d_in[23];
    const float* p2_m1b = (const float*)d_in[24];
    const float* p2_m2w = (const float*)d_in[25];
    const float* p2_m2b = (const float*)d_in[26];
    const float* projW  = (const float*)d_in[27];
    const float* projB  = (const float*)d_in[28];

    const int B = 64;
    const int T1 = 56 * 56, BT1 = B * T1;     // 200704
    const int T2 = 28 * 28, BT2 = B * T2;     // 50176
    const int T3 = 14 * 14, BT3 = B * T3;     // 12544
    const int d1 = 147, d1p = 160;
    const int d2 = 576;
    const int Tp1 = 3136;                     // T1, already mult of 32
    const int Tp2 = 800;                      // T2=784 padded to mult of 32

    char* ws = (char*)d_ws;
    size_t off = 0;
    auto carve = [&](size_t bytes) -> char* {
        char* p = ws + off;
        off += (bytes + 255) & ~(size_t)255;
        return p;
    };
    // fp16 activation rows and temps
    _Float16* A16   = (_Float16*)carve((size_t)BT1 * d1p * sizeof(_Float16)); // >= BT2*d2
    float*    KQV   = (float*)   carve((size_t)BT1 * 192 * sizeof(float));
    float*    KP    = (float*)   carve((size_t)BT1 * 32  * sizeof(float));
    float*    QP    = (float*)   carve((size_t)BT1 * 32  * sizeof(float));
    float*    KPS   = (float*)   carve((size_t)B   * 32  * sizeof(float));
    float*    KPTV  = (float*)   carve((size_t)B * 64 * 32 * sizeof(float));
    float*    Ybuf  = (float*)   carve((size_t)BT1 * 64 * sizeof(float));
    _Float16* T16a  = (_Float16*)carve((size_t)BT1 * 64 * sizeof(_Float16));
    _Float16* T16b  = (_Float16*)carve((size_t)BT1 * 64 * sizeof(_Float16));
    float*    OutTok= (float*)   carve((size_t)BT1 * 64 * sizeof(float));
    // packed performer operands
    _Float16* VT16  = (_Float16*)carve((size_t)B * 64 * Tp1 * sizeof(_Float16));
    _Float16* KPT16 = (_Float16*)carve((size_t)B * 32 * Tp1 * sizeof(_Float16));
    // preconverted fp16 weights (zero-padded to KdPad)
    _Float16* W1kqv = (_Float16*)carve((size_t)192 * d1p * sizeof(_Float16));
    _Float16* W1pr  = (_Float16*)carve((size_t)64 * 64 * sizeof(_Float16));
    _Float16* W1m1  = (_Float16*)carve((size_t)64 * 64 * sizeof(_Float16));
    _Float16* W1m2  = (_Float16*)carve((size_t)64 * 64 * sizeof(_Float16));
    _Float16* W2kqv = (_Float16*)carve((size_t)192 * d2 * sizeof(_Float16));
    _Float16* W2pr  = (_Float16*)carve((size_t)64 * 64 * sizeof(_Float16));
    _Float16* W2m1  = (_Float16*)carve((size_t)64 * 64 * sizeof(_Float16));
    _Float16* W2m2  = (_Float16*)carve((size_t)64 * 64 * sizeof(_Float16));
    _Float16* Wproj = (_Float16*)carve((size_t)768 * d2 * sizeof(_Float16));

    auto cvtw = [&](const float* W, int N, int Kd, int KdPad, _Float16* W16) {
        int total = N * KdPad;
        convert_w_kernel<<<(total + 255) / 256, 256, 0, stream>>>(W, N, Kd, KdPad, W16);
    };
    cvtw(p1_kqvw, 192, d1, d1p, W1kqv);
    cvtw(p1_prw,  64, 64, 64, W1pr);
    cvtw(p1_m1w,  64, 64, 64, W1m1);
    cvtw(p1_m2w,  64, 64, 64, W1m2);
    cvtw(p2_kqvw, 192, d2, d2, W2kqv);
    cvtw(p2_prw,  64, 64, 64, W2pr);
    cvtw(p2_m1w,  64, 64, 64, W2m1);
    cvtw(p2_m2w,  64, 64, 64, W2m2);
    cvtw(projW,  768, d2, d2, Wproj);

    // ---------------- Stage 1 (7x7 s4 p2 on 3x224x224) ----------------
    unfold_ln_kernel<<<BT1, 32, 0, stream>>>(x, 0, 3, 224, 224, 7, 4, 2,
                                             56, 56, d1, d1p, p1_n1g, p1_n1b, 1, A16);
    wmma_gemm_kernel<4><<<dim3(BT1/16, 192/64), 32, 0, stream>>>(
        A16, d1p, W1kqv, p1_kqvb, nullptr, 0, 0, KQV, nullptr, 192, 0);
    prm_exp_kernel<<<BT1, 32, 0, stream>>>(KQV, p1_w, KP, QP);
    kps_kernel<<<B, 256, 0, stream>>>(KP, T1, KPS);
    pack_vt_kernel <<<(B*64*Tp1 + 255)/256, 256, 0, stream>>>(KQV, T1, Tp1, VT16);
    pack_kpt_kernel<<<(B*32*Tp1 + 255)/256, 256, 0, stream>>>(KP, T1, Tp1, KPT16);
    kptv_kernel<<<B, 256, 0, stream>>>(VT16, KPT16, Tp1, KPTV);
    yattn_kernel<<<BT1, 32, 0, stream>>>(QP, KPS, KPTV, T1, T16a);
    wmma_gemm_kernel<4><<<dim3(BT1/16, 1), 32, 0, stream>>>(
        T16a, 64, W1pr, p1_prb, KQV, 192, 128, Ybuf, nullptr, 64, 0);   // y = v + proj
    ln64_kernel<<<BT1, 32, 0, stream>>>(Ybuf, p1_n2g, p1_n2b, T16a);
    wmma_gemm_kernel<4><<<dim3(BT1/16, 1), 32, 0, stream>>>(
        T16a, 64, W1m1, p1_m1b, nullptr, 0, 0, nullptr, T16b, 64, 1);   // gelu
    wmma_gemm_kernel<4><<<dim3(BT1/16, 1), 32, 0, stream>>>(
        T16b, 64, W1m2, p1_m2b, Ybuf, 64, 0, OutTok, nullptr, 64, 0);   // y + h

    // ---------------- Stage 2 (3x3 s2 p1 on 64x56x56) ----------------
    unfold_ln_kernel<<<BT2, 32, 0, stream>>>(OutTok, 1, 64, 56, 56, 3, 2, 1,
                                             28, 28, d2, d2, p2_n1g, p2_n1b, 1, A16);
    wmma_gemm_kernel<4><<<dim3(BT2/16, 192/64), 32, 0, stream>>>(
        A16, d2, W2kqv, p2_kqvb, nullptr, 0, 0, KQV, nullptr, 192, 0);
    prm_exp_kernel<<<BT2, 32, 0, stream>>>(KQV, p2_w, KP, QP);
    kps_kernel<<<B, 256, 0, stream>>>(KP, T2, KPS);
    pack_vt_kernel <<<(B*64*Tp2 + 255)/256, 256, 0, stream>>>(KQV, T2, Tp2, VT16);
    pack_kpt_kernel<<<(B*32*Tp2 + 255)/256, 256, 0, stream>>>(KP, T2, Tp2, KPT16);
    kptv_kernel<<<B, 256, 0, stream>>>(VT16, KPT16, Tp2, KPTV);
    yattn_kernel<<<BT2, 32, 0, stream>>>(QP, KPS, KPTV, T2, T16a);
    wmma_gemm_kernel<4><<<dim3(BT2/16, 1), 32, 0, stream>>>(
        T16a, 64, W2pr, p2_prb, KQV, 192, 128, Ybuf, nullptr, 64, 0);
    ln64_kernel<<<BT2, 32, 0, stream>>>(Ybuf, p2_n2g, p2_n2b, T16a);
    wmma_gemm_kernel<4><<<dim3(BT2/16, 1), 32, 0, stream>>>(
        T16a, 64, W2m1, p2_m1b, nullptr, 0, 0, nullptr, T16b, 64, 1);
    wmma_gemm_kernel<4><<<dim3(BT2/16, 1), 32, 0, stream>>>(
        T16b, 64, W2m2, p2_m2b, Ybuf, 64, 0, OutTok, nullptr, 64, 0);

    // ---------------- Stage 3 (3x3 s2 p1 on 64x28x28) + 576->768 ----------
    unfold_ln_kernel<<<BT3, 32, 0, stream>>>(OutTok, 1, 64, 28, 28, 3, 2, 1,
                                             14, 14, d2, d2, nullptr, nullptr, 0, A16);
    wmma_gemm_kernel<4><<<dim3(BT3/16, 768/64), 32, 0, stream>>>(
        A16, d2, Wproj, projB, nullptr, 0, 0, (float*)d_out, nullptr, 768, 0);
}
